// SimVQ1D_23819888623709
// MI455X (gfx1250) — matrix operand order repaired and checked by reading, not compiled, not run on hardware
//
#include <hip/hip_runtime.h>

// ---------------------------------------------------------------------------
// SimVQ forward for MI455X (gfx1250, wave32).
// Dominant cost: 32768x65536x32 distance GEMM + fused argmin (~137 GFLOP).
// argmin(dist) == argmax(z.cb - ||cb||^2/2); fold -||cb||^2/2 into the WMMA
// f32 accumulator; z.cb via v_wmma_f32_16x16x32_bf16 (K=32 == D, one WMMA per
// 16x16 tile). Codebook (4MB bf16) is L2-resident; staged through LDS.
// K-dimension split 4 ways across blockIdx.y for wave parallelism (4096 waves)
// without increasing codebook traffic; partial argmax combined in kquant.
// ---------------------------------------------------------------------------

#define K_CODES 65536
#define NTOK    32768          // B*T = 8*4096
#define EPSF    1e-8f
#define NOISE_STD 0.0316227766016838f   // sqrt(0.001)
#define CPI     128            // codes staged in LDS per chunk
#define SPLIT   4              // codebook scan split across blockIdx.y

typedef __attribute__((ext_vector_type(16))) __bf16 v16bf;
typedef __attribute__((ext_vector_type(8)))  float  v8f;

union FragAB { v16bf v; unsigned int u[8]; uint4 q[2]; };
union FragC  { v8f   v; float f[8]; };

__device__ __forceinline__ unsigned int pack_bf16x2(float lo, float hi) {
    unsigned int ulo = __float_as_uint(lo);
    unsigned int uhi = __float_as_uint(hi);
    ulo += 0x7FFFu + ((ulo >> 16) & 1u);   // round-to-nearest-even
    uhi += 0x7FFFu + ((uhi >> 16) & 1u);
    return (ulo >> 16) | (uhi & 0xFFFF0000u);
}

// ---------------------------------------------------------------------------
// Kernel A: cb = codebook @ W + bias; emit f32 copy, bf16 copy, -0.5*||cb||^2,
// and zero the counts buffer (must re-zero every launch for graph replay).
// One 32-lane group per codebook row; lane == output dim.
// ---------------------------------------------------------------------------
__global__ __launch_bounds__(256) void kproj(const float* __restrict__ codebook,
                                             const float* __restrict__ W,
                                             const float* __restrict__ bias,
                                             float* __restrict__ cbf,
                                             unsigned short* __restrict__ cbh,
                                             float* __restrict__ negHalfNorm,
                                             float* __restrict__ counts) {
    __shared__ float sW[32 * 32];
    __shared__ float sB[32];
    const int tid = threadIdx.x;
    for (int i = tid; i < 1024; i += 256) sW[i] = W[i];
    if (tid < 32) sB[tid] = bias[tid];
    __syncthreads();

    const int row = blockIdx.x * 8 + (tid >> 5);
    const int d   = tid & 31;
    const float* crow = codebook + row * 32;
    float acc = sB[d];
#pragma unroll
    for (int c = 0; c < 32; ++c) acc = fmaf(crow[c], sW[c * 32 + d], acc);

    cbf[row * 32 + d] = acc;
    unsigned int u = __float_as_uint(acc);
    u += 0x7FFFu + ((u >> 16) & 1u);
    cbh[row * 32 + d] = (unsigned short)(u >> 16);

    float sq = acc * acc;
#pragma unroll
    for (int m = 16; m >= 1; m >>= 1) sq += __shfl_xor(sq, m, 32);
    if (d == 0) negHalfNorm[row] = -0.5f * sq;
    if (d == 1) counts[row] = 0.0f;
}

// ---------------------------------------------------------------------------
// Kernel B: fused GEMM + argmax. 8 waves/WG, 32 tokens/wave (2 A tiles),
// 256 tokens/WG. blockIdx.y selects a 16K-code quarter of the codebook.
// Codebook quarter streamed through LDS in 128-code chunks.
// D-tile layout (wave32, f32 16x16): lane holds 8 tokens x 1 code, so the
// argmax candidate index is uniform per lane; final argmax is a 16-lane
// shfl_xor tree. Per-quarter (score, idx) written for later combine.
// ---------------------------------------------------------------------------
__global__ __launch_bounds__(256) void ksearch(const float* __restrict__ z,
                                               const uint4* __restrict__ cbh4,
                                               const float* __restrict__ negHalfNorm,
                                               float* __restrict__ partScore,
                                               int* __restrict__ partIdx) {
    __shared__ uint4 sCodes[CPI * 4];   // 128 codes * 64B (32 bf16)
    __shared__ float sEn[CPI];          // pre-scaled -0.5*||cb||^2

    const int tid  = threadIdx.x;
    const int lane = tid & 31;
    const int wid  = tid >> 5;
    const int half = lane >> 4;   // 0: k-lo half, 1: k-hi half
    const int l15  = lane & 15;

    const int hsplit = blockIdx.y;
    const int jstart = hsplit * (K_CODES / SPLIT);
    const int jend   = jstart + (K_CODES / SPLIT);

    // Build two 16x32 bf16 A fragments (token tiles) from f32 z.
    // Lane l (l<16): row m=l, k in [0..7]U[16..23]; lane l+16: k in [8..15]U[24..31].
    const int tile0 = blockIdx.x * 256 + wid * 32;
    FragAB a0, a1;
    {
        const int koff = half ? 8 : 0;
#pragma unroll
        for (int t = 0; t < 2; ++t) {
            const float* zr = z + (size_t)(tile0 + t * 16 + l15) * 32 + koff;
            FragAB& A = t ? a1 : a0;
#pragma unroll
            for (int h = 0; h < 2; ++h) {            // h: k+0.. / k+16..
                const float4 p0 = ((const float4*)(zr + h * 16))[0];
                const float4 p1 = ((const float4*)(zr + h * 16))[1];
                A.u[h * 4 + 0] = pack_bf16x2(p0.x, p0.y);
                A.u[h * 4 + 1] = pack_bf16x2(p0.z, p0.w);
                A.u[h * 4 + 2] = pack_bf16x2(p1.x, p1.y);
                A.u[h * 4 + 3] = pack_bf16x2(p1.z, p1.w);
            }
        }
    }

    float bs0[8], bs1[8];
    int   bi0[8], bi1[8];
#pragma unroll
    for (int v = 0; v < 8; ++v) { bs0[v] = bs1[v] = -3.4e38f; bi0[v] = bi1[v] = 0; }

    for (int jb = jstart; jb < jend; jb += CPI) {
        // Cooperative stage: 8KB codes + 512B norms into LDS.
        sCodes[tid * 2]     = cbh4[(size_t)jb * 4 + tid * 2];
        sCodes[tid * 2 + 1] = cbh4[(size_t)jb * 4 + tid * 2 + 1];
        if (tid < CPI) sEn[tid] = negHalfNorm[jb + tid];
        if (jb + CPI < jend)   // prefetch next chunk (global_prefetch_b8)
            __builtin_prefetch((const char*)(cbh4 + (size_t)(jb + CPI) * 4) + tid * 32, 0, 0);
        __syncthreads();

#pragma unroll
        for (int c = 0; c < CPI; c += 16) {
            // B fragment (32x16 bf16): lane holds code row (c+l15), k-half by lane half.
            FragAB b;
            const int base = (c + l15) * 4 + half;
            b.q[0] = sCodes[base];
            b.q[1] = sCodes[base + 2];

            const float en = sEn[c + l15];
            FragC cinit;
#pragma unroll
            for (int v = 0; v < 8; ++v) cinit.f[v] = en;   // C = -||cb_j||^2/2

            const int cand = jb + c + l15;                 // uniform per lane
            FragC r0, r1;
            r0.v = __builtin_amdgcn_wmma_f32_16x16x32_bf16(
                false, a0.v, false, b.v, (short)0, cinit.v, false, false);
            r1.v = __builtin_amdgcn_wmma_f32_16x16x32_bf16(
                false, a1.v, false, b.v, (short)0, cinit.v, false, false);
#pragma unroll
            for (int v = 0; v < 8; ++v) {
                bool g0 = r0.f[v] > bs0[v];
                bs0[v] = g0 ? r0.f[v] : bs0[v];
                bi0[v] = g0 ? cand    : bi0[v];
                bool g1 = r1.f[v] > bs1[v];
                bs1[v] = g1 ? r1.f[v] : bs1[v];
                bi1[v] = g1 ? cand    : bi1[v];
            }
        }
        __syncthreads();
    }

    // Argmax across the 16 lanes that share each token (ties -> smaller index,
    // matching jnp.argmin first-hit semantics).
#pragma unroll
    for (int v = 0; v < 8; ++v) {
        float s0 = bs0[v]; int i0 = bi0[v];
        float s1 = bs1[v]; int i1 = bi1[v];
#pragma unroll
        for (int m = 1; m < 16; m <<= 1) {
            float so = __shfl_xor(s0, m, 32); int io = __shfl_xor(i0, m, 32);
            if (so > s0 || (so == s0 && io < i0)) { s0 = so; i0 = io; }
            so = __shfl_xor(s1, m, 32); io = __shfl_xor(i1, m, 32);
            if (so > s1 || (so == s1 && io < i1)) { s1 = so; i1 = io; }
        }
        if (l15 == 0) {
            const int m = v + half * 8;
            partScore[hsplit * NTOK + tile0 + m]      = s0;
            partIdx  [hsplit * NTOK + tile0 + m]      = i0;
            partScore[hsplit * NTOK + tile0 + 16 + m] = s1;
            partIdx  [hsplit * NTOK + tile0 + 16 + m] = i1;
        }
    }
}

// ---------------------------------------------------------------------------
// Kernel C: combine the SPLIT partial argmax results, then noisy
// quantization. One 32-lane group per token, lane == dim. Also does the
// histogram atomics and writes idx as float to the output tuple.
// Quarters are in ascending index order, so strict > keeps the smaller index
// on score ties (first-min semantics).
// ---------------------------------------------------------------------------
__global__ __launch_bounds__(256) void kquant(const float* __restrict__ z,
                                              const float* __restrict__ noise,
                                              const float* __restrict__ cbf,
                                              const float* __restrict__ partScore,
                                              const int* __restrict__ partIdx,
                                              float* __restrict__ zq,
                                              float* __restrict__ idxOutF,
                                              float* __restrict__ counts) {
    const int tid = threadIdx.x;
    const int t   = blockIdx.x * 8 + (tid >> 5);
    const int d   = tid & 31;

    float bs = partScore[t];
    int   id = partIdx[t];
#pragma unroll
    for (int h = 1; h < SPLIT; ++h) {
        const float s = partScore[h * NTOK + t];
        const int   i = partIdx[h * NTOK + t];
        const bool  g = s > bs;
        bs = g ? s : bs;
        id = g ? i : id;
    }

    const float zd = z[(size_t)t * 32 + d];
    const float qd = cbf[(size_t)id * 32 + d];
    const float nd = noise[(size_t)t * 32 + d];

    const float diff = qd - zd;
    float s = diff * diff;
#pragma unroll
    for (int m = 16; m >= 1; m >>= 1) s += __shfl_xor(s, m, 32);
    const float diff_norm = fmaxf(sqrtf(s), EPSF);

    const float dd = NOISE_STD * nd + diff;
    float s2 = dd * dd;
#pragma unroll
    for (int m = 16; m >= 1; m >>= 1) s2 += __shfl_xor(s2, m, 32);
    const float dir_norm = fmaxf(sqrtf(s2), EPSF);

    zq[(size_t)t * 32 + d] = zd + diff_norm * (dd / dir_norm);

    if (d == 0) {
        atomicAdd(&counts[id], 1.0f);
        idxOutF[t] = (float)id;
    }
}

// ---------------------------------------------------------------------------
// Kernel D: perplexity + usage ratio over K=65536 counts. Single block.
// ---------------------------------------------------------------------------
__global__ __launch_bounds__(256) void kstats(const float* __restrict__ counts,
                                              float* __restrict__ outPerp,
                                              float* __restrict__ outUsage) {
    __shared__ float red[256];
    const int tid = threadIdx.x;

    float sc = 0.f;
    for (int k = tid; k < K_CODES; k += 256) sc += counts[k];
    red[tid] = sc; __syncthreads();
    for (int s = 128; s > 0; s >>= 1) { if (tid < s) red[tid] += red[tid + s]; __syncthreads(); }
    const float total = fmaxf(red[0], 1.0f);
    __syncthreads();

    float ent = 0.f, nz = 0.f;
    for (int k = tid; k < K_CODES; k += 256) {
        const float c = counts[k];
        const float p = c / total;
        const float sp = (p > 0.f) ? p : 1.0f;
        ent += p * logf(sp + 1e-10f);
        nz  += (c > 0.f) ? 1.f : 0.f;
    }
    red[tid] = ent; __syncthreads();
    for (int s = 128; s > 0; s >>= 1) { if (tid < s) red[tid] += red[tid + s]; __syncthreads(); }
    const float entTot = red[0];
    __syncthreads();
    red[tid] = nz; __syncthreads();
    for (int s = 128; s > 0; s >>= 1) { if (tid < s) red[tid] += red[tid + s]; __syncthreads(); }
    if (tid == 0) {
        *outPerp  = expf(-entTot);
        *outUsage = red[0] / (float)K_CODES;
    }
}

// ---------------------------------------------------------------------------
// d_in order: z, codebook, W, proj_bias, noise.
// d_out: z_q (1048576 f32) | perplexity (1) | usage (1) | idx-as-float (32768).
// ---------------------------------------------------------------------------
extern "C" void kernel_launch(void* const* d_in, const int* in_sizes, int n_in,
                              void* d_out, int out_size, void* d_ws, size_t ws_size,
                              hipStream_t stream) {
    (void)in_sizes; (void)n_in; (void)out_size; (void)ws_size;
    const float* z        = (const float*)d_in[0];
    const float* codebook = (const float*)d_in[1];
    const float* W        = (const float*)d_in[2];
    const float* bias     = (const float*)d_in[3];
    const float* noise    = (const float*)d_in[4];

    char* ws = (char*)d_ws;
    float*          cbf         = (float*)ws;                        //  8 MB f32 projected cb
    unsigned short* cbh         = (unsigned short*)(ws + 8388608);   //  4 MB bf16 cb
    float*          negHalfNorm = (float*)(ws + 12582912);           // 256 KB
    float*          partScore   = (float*)(ws + 12845056);           // 512 KB (SPLIT * NTOK)
    int*            partIdx     = (int*)(ws + 13369344);             // 512 KB
    float*          counts      = (float*)(ws + 13893632);           // 256 KB

    float* outZq    = (float*)d_out;
    float* outPerp  = outZq + 1048576;
    float* outUsage = outZq + 1048577;
    float* outIdxF  = outZq + 1048578;

    kproj  <<<K_CODES / 8, 256, 0, stream>>>(codebook, W, bias, cbf, cbh, negHalfNorm, counts);
    ksearch<<<dim3(NTOK / 256, SPLIT), 256, 0, stream>>>(z, (const uint4*)cbh, negHalfNorm,
                                                         partScore, partIdx);
    kquant <<<NTOK / 8, 256, 0, stream>>>(z, noise, cbf, partScore, partIdx,
                                          outZq, outIdxF, counts);
    kstats <<<1, 256, 0, stream>>>(counts, outPerp, outUsage);
}
